// BBAstar_39608188404231
// MI455X (gfx1250) — matrix operand order repaired and checked by reading, compiled, not run on hardware
//
#include <hip/hip_runtime.h>
#include <math.h>

typedef __attribute__((ext_vector_type(16))) _Float16 v16h;
typedef __attribute__((ext_vector_type(8)))  _Float16 v8h;
typedef __attribute__((ext_vector_type(8)))  float    v8f;

#define NB 32
#define HWFULL 4096

static __host__ inline int ceil_div(int a, int b) { return (a + b - 1) / b; }

// ---------------------------------------------------------------------------
// Input prep: x = concat([map, start+goal], C) in NHWC fp16 (32,64,64,2)
// ---------------------------------------------------------------------------
__global__ __launch_bounds__(256) void prep_input_k(
    const float* __restrict__ md, const float* __restrict__ st,
    const float* __restrict__ gl, _Float16* __restrict__ x) {
  int i = blockIdx.x * 256 + threadIdx.x;
  if (i >= NB * HWFULL) return;
  x[2 * i + 0] = (_Float16)md[i];
  x[2 * i + 1] = (_Float16)(st[i] + gl[i]);
}

// ---------------------------------------------------------------------------
// Weight transform: OIHW f32 -> [Cout][Kpad] f16, k = (kh*3+kw)*Cin + ci
// ---------------------------------------------------------------------------
__global__ __launch_bounds__(256) void xform_w_k(
    const float* __restrict__ w, _Float16* __restrict__ wt,
    int Cin, int Cout, int K, int Kpad) {
  int i = blockIdx.x * 256 + threadIdx.x;
  if (i >= Cout * Kpad) return;
  int co = i / Kpad;
  int k  = i - co * Kpad;
  float v = 0.0f;
  if (k < K) {
    int tap = k / Cin;
    int ci  = k - tap * Cin;
    v = w[(size_t)(co * Cin + ci) * 9 + tap];
  }
  wt[i] = (_Float16)v;
}

// ---------------------------------------------------------------------------
// Implicit-GEMM conv + BN-scale/bias + ReLU via V_WMMA_F32_16X16X32_F16.
// Block = 8 waves; wave w owns mtile = blockIdx.x*8 + w (16 spatial rows).
// blockIdx.y selects a group of NT n-tiles (NT*16 output channels).
// Per 32-K step: weights tile (NT*16 cols x 32 k) staged to LDS with
// global_load_async_to_lds_b128 (ASYNCcnt), then each wave runs NT WMMAs
// off one A fragment.  in: NHWC f16; wt: [Cout][Kpad] f16; out: NHWC f16.
// ---------------------------------------------------------------------------
template <int NT>
__global__ __launch_bounds__(256) void conv_wmma_k(
    const _Float16* __restrict__ in, const _Float16* __restrict__ wt,
    const float* __restrict__ sc, const float* __restrict__ bi,
    _Float16* __restrict__ out,
    int H, int W, int logW, int logHW, int Cin, int logCin,
    int Cout, int K, int Kpad, int relu) {
  __shared__ __align__(64) _Float16 ldsB[64 * 32];  // [colLocal][k], 4KB

  int tid  = threadIdx.x;
  int wave = tid >> 5, lane = tid & 31;
  int mtile   = blockIdx.x * 8 + wave;
  int co_base = blockIdx.y * (NT * 16);

  int half  = lane >> 4;
  int hwm1  = (1 << logHW) - 1;
  int mrowA = (mtile << 4) + (lane & 15);  // A-matrix row for this lane
  int n_img = mrowA >> logHW;
  int rem   = mrowA & hwm1;
  int oh    = rem >> logW;
  int ow    = rem & (W - 1);
  int vecCin = ((Cin & 7) == 0);

  // cooperative B staging: thread -> (colLocal = tid/4, 8-half chunk tid%4)
  int col  = tid >> 2;        // 0..63
  int koff = (tid & 3) << 3;  // 0,8,16,24
  unsigned ldsAddr = (unsigned)(size_t)&ldsB[(col << 5) + koff];
  const _Float16* gB = wt + (size_t)(co_base + col) * Kpad + koff;
  bool doCopy = (col < NT * 16);

  v8f acc[NT];
#pragma unroll
  for (int j = 0; j < NT; ++j) acc[j] = {};

  for (int k0 = 0; k0 < K; k0 += 32) {
    // async 16B/lane copy of the 32K x NT*16 weight tile into LDS
    if (doCopy) {
      unsigned long long ga = (unsigned long long)(gB + k0);
      asm volatile("global_load_async_to_lds_b128 %0, %1, off"
                   :: "v"(ldsAddr), "v"(ga) : "memory");
    }
    asm volatile("s_wait_asynccnt 0" ::: "memory");
    __syncthreads();

    // A fragment (ISA 16-bit A 16x32 layout)
    v16h a;
    if (vecCin) {
#pragma unroll
      for (int g2 = 0; g2 < 2; ++g2) {
        int kb  = k0 + (g2 << 4) + (half << 3);
        int tap = kb >> logCin;
        int ci  = kb & (Cin - 1);
        int kh  = tap / 3, kw = tap - kh * 3;
        int ih  = oh + kh - 1, iw = ow + kw - 1;
        v8h chunk = {};
        if ((unsigned)ih < (unsigned)H && (unsigned)iw < (unsigned)W) {
          size_t idx = ((size_t)((n_img * H + ih) * W + iw)) * Cin + ci;
          chunk = *(const v8h*)(in + idx);
        }
#pragma unroll
        for (int j = 0; j < 8; ++j) a[(g2 << 3) + j] = chunk[j];
      }
    } else {
      // scalar fallback (first conv, Cin=2, K=18 zero-padded to 32)
#pragma unroll
      for (int j = 0; j < 16; ++j) {
        int k = k0 + ((j >> 3) << 4) + (half << 3) + (j & 7);
        _Float16 v = (_Float16)0.0f;
        if (k < K) {
          int tap = k / Cin;
          int ci  = k - tap * Cin;
          int kh  = tap / 3, kw = tap - kh * 3;
          int ih  = oh + kh - 1, iw = ow + kw - 1;
          if ((unsigned)ih < (unsigned)H && (unsigned)iw < (unsigned)W)
            v = in[((size_t)((n_img * H + ih) * W + iw)) * Cin + ci];
        }
        a[j] = v;
      }
    }

#pragma unroll
    for (int j = 0; j < NT; ++j) {
      const _Float16* bp = &ldsB[(((j << 4) + (lane & 15)) << 5) + (half << 3)];
      v8h b0 = *(const v8h*)bp, b1 = *(const v8h*)(bp + 16);
      v16h bm = __builtin_shufflevector(b0, b1, 0, 1, 2, 3, 4, 5, 6, 7, 8, 9, 10, 11, 12, 13, 14, 15);
      acc[j] = __builtin_amdgcn_wmma_f32_16x16x32_f16(
          false, a, false, bm, (short)0, acc[j], false, false);
    }
    __syncthreads();  // ds reads retired (wmma waited on DScnt) before restage
  }

  // Epilogue: D layout -> m = r + 8*half, n = lane&15
  int mbase = (mtile << 4) + (half << 3);
#pragma unroll
  for (int j = 0; j < NT; ++j) {
    int co = co_base + (j << 4) + (lane & 15);
    float s = sc[co], bb = bi[co];
#pragma unroll
    for (int r = 0; r < 8; ++r) {
      int m   = mbase + r;
      int n2  = m >> logHW;
      int rm  = m & hwm1;
      int oh2 = rm >> logW, ow2 = rm & (W - 1);
      float v = acc[j][r] * s + bb;
      if (relu) v = v > 0.0f ? v : 0.0f;
      out[((size_t)((n2 * H + oh2) * W + ow2)) * Cout + co] = (_Float16)v;
    }
  }
}

// ---------------------------------------------------------------------------
// 2x2 maxpool, NHWC f16
// ---------------------------------------------------------------------------
__global__ __launch_bounds__(256) void maxpool2_k(
    const _Float16* __restrict__ in, _Float16* __restrict__ out, int H, int C) {
  int Ho = H >> 1;
  int total = NB * Ho * Ho * C;
  int i = blockIdx.x * 256 + threadIdx.x;
  if (i >= total) return;
  int c = i % C; int t = i / C;
  int ow = t % Ho; t /= Ho;
  int oh = t % Ho; int n = t / Ho;
  float m = -3.4e38f;
#pragma unroll
  for (int dy = 0; dy < 2; ++dy)
#pragma unroll
    for (int dx = 0; dx < 2; ++dx) {
      float v = (float)in[((size_t)((n * H + (2 * oh + dy)) * H + (2 * ow + dx))) * C + c];
      m = v > m ? v : m;
    }
  out[i] = (_Float16)m;
}

// ---------------------------------------------------------------------------
// up2 (nearest 2x) of `a` + channel-concat with `skip` (C2==0 -> no skip)
// ---------------------------------------------------------------------------
__global__ __launch_bounds__(256) void up2cat_k(
    const _Float16* __restrict__ a, const _Float16* __restrict__ skip,
    _Float16* __restrict__ out, int Ho, int C1, int C2) {
  int C = C1 + C2;
  int total = NB * Ho * Ho * C;
  int i = blockIdx.x * 256 + threadIdx.x;
  if (i >= total) return;
  int c = i % C; int t = i / C;
  int w = t % Ho; t /= Ho;
  int h = t % Ho; int n = t / Ho;
  if (c < C1) {
    int Hi = Ho >> 1;
    out[i] = a[((size_t)((n * Hi + (h >> 1)) * Hi + (w >> 1))) * C1 + c];
  } else {
    out[i] = skip[((size_t)((n * Ho + h) * Ho + w)) * C2 + (c - C1)];
  }
}

// ---------------------------------------------------------------------------
// Head: 3x3 conv (32->1) f32 + sigmoid -> cost (ws f32 + d_out section)
// ---------------------------------------------------------------------------
__global__ __launch_bounds__(256) void head_k(
    const _Float16* __restrict__ in, const float* __restrict__ w,
    const float* __restrict__ b, float* __restrict__ cost_ws,
    float* __restrict__ cost_out) {
  int i = blockIdx.x * 256 + threadIdx.x;
  if (i >= NB * HWFULL) return;
  int n = i >> 12; int cell = i & 4095;
  int y = cell >> 6, x = cell & 63;
  float acc = b[0];
#pragma unroll
  for (int kh = 0; kh < 3; ++kh)
#pragma unroll
    for (int kw = 0; kw < 3; ++kw) {
      int ih = y + kh - 1, iw = x + kw - 1;
      if ((unsigned)ih < 64u && (unsigned)iw < 64u) {
        const _Float16* px = in + ((size_t)((n * 64 + ih) * 64 + iw)) * 32;
        int tap = kh * 3 + kw;
        for (int ci = 0; ci < 32; ++ci) acc += (float)px[ci] * w[ci * 9 + tap];
      }
    }
  float sg = 1.0f / (1.0f + expf(-acc));
  cost_ws[i] = sg;
  cost_out[i] = sg;
}

// ---------------------------------------------------------------------------
// A* prep kernels
// ---------------------------------------------------------------------------
__global__ __launch_bounds__(256) void find_goal_k(const float* __restrict__ goal,
                                                   int* __restrict__ gidx) {
  int i = blockIdx.x * 256 + threadIdx.x;
  if (i >= NB * HWFULL) return;
  if (goal[i] > 0.5f) gidx[i >> 12] = i & 4095;
}

__global__ __launch_bounds__(256) void compute_h_k(const float* __restrict__ cost,
                                                   const int* __restrict__ gidx,
                                                   float* __restrict__ hout) {
  int i = blockIdx.x * 256 + threadIdx.x;
  if (i >= NB * HWFULL) return;
  int n = i >> 12; int cell = i & 4095;
  int y = cell >> 6, x = cell & 63;
  int gi = gidx[n]; int gy = gi >> 6, gx = gi & 63;
  float dy = fabsf((float)(y - gy)), dx = fabsf((float)(x - gx));
  float cheb = dy + dx - fminf(dy, dx);
  float euc = sqrtf(dy * dy + dx * dx);
  hout[i] = cheb + 0.001f * euc + cost[i];
}

__global__ __launch_bounds__(256) void init_astar_k(
    const float* __restrict__ start, const int* __restrict__ gidx,
    float* __restrict__ op, float* __restrict__ hist, float* __restrict__ g,
    int* __restrict__ par) {
  int i = blockIdx.x * 256 + threadIdx.x;
  if (i >= NB * HWFULL) return;
  op[i] = start[i];
  hist[i] = 0.0f;
  g[i] = 0.0f;
  par[i] = gidx[i >> 12];
}

// ---------------------------------------------------------------------------
// A* main loop: one workgroup (256 thr = 8 waves) per batch item; Tmax serial
// steps of (argmin f over open cells via LDS tree reduce; expand 8 neighbors),
// then 1024-step backtrack on thread 0.
// ---------------------------------------------------------------------------
__global__ __launch_bounds__(256) void astar_k(
    const float* __restrict__ cost, const float* __restrict__ hmap,
    const float* __restrict__ obst, const float* __restrict__ goal,
    const int* __restrict__ gidx,
    float* __restrict__ open_, float* __restrict__ hist_,
    float* __restrict__ g_, int* __restrict__ par_,
    float* __restrict__ out_hist, float* __restrict__ out_path, int Tmax) {
  int n = blockIdx.x;
  int tid = threadIdx.x;
  const int HW = HWFULL;
  float* op = open_ + n * HW;
  float* hi = hist_ + n * HW;
  float* gg = g_ + n * HW;
  int*   pp = par_ + n * HW;
  const float* hn = hmap + (size_t)n * HW;
  const float* cn = cost + (size_t)n * HW;
  const float* ob = obst + (size_t)n * HW;

  __shared__ float sf[256];
  __shared__ int   si[256];
  __shared__ int   s_ind;
  __shared__ float s_g2;
  __shared__ int   s_goal;
  if (tid == 0) s_goal = gidx[n];
  __syncthreads();

  for (int t = 0; t < Tmax; ++t) {
    // argmin of f = 0.5*g + 0.5*h over open cells, lowest-index tie-break
    float bf = 3.4e38f;
    int bidx = 1 << 30;
    for (int c = 0; c < 16; ++c) {
      int i = (c << 8) + tid;
      if (op[i] > 0.5f) {
        float f = 0.5f * gg[i] + 0.5f * hn[i];
        if (f < bf || (f == bf && i < bidx)) { bf = f; bidx = i; }
      }
    }
    sf[tid] = bf; si[tid] = bidx;
    __syncthreads();
    for (int s2 = 128; s2 > 0; s2 >>= 1) {
      if (tid < s2) {
        float f2 = sf[tid + s2]; int i2 = si[tid + s2];
        if (f2 < sf[tid] || (f2 == sf[tid] && i2 < si[tid])) { sf[tid] = f2; si[tid] = i2; }
      }
      __syncthreads();
    }
    if (tid == 0) {
      int ind = si[0];
      if (ind >= HW) ind = 0;  // no open cells -> argmax(zeros) = 0
      float unsolved = (ind == s_goal) ? 0.0f : 1.0f;
      hi[ind] = 1.0f;
      op[ind] = fmaxf(op[ind] - unsolved, 0.0f);
      s_g2 = gg[ind] + cn[ind];
      s_ind = ind;
    }
    __syncthreads();
    if (tid < 8) {
      const int dy[8] = {-1, -1, -1, 0, 0, 1, 1, 1};
      const int dx[8] = {-1, 0, 1, -1, 1, -1, 0, 1};
      int ind = s_ind;
      int y = ind >> 6, x = ind & 63;
      int ny = y + dy[tid], nx = x + dx[tid];
      if ((unsigned)ny < 64u && (unsigned)nx < 64u) {
        int v = (ny << 6) + nx;
        float nb = ob[v];
        if (nb > 0.5f) {
          float ov = op[v], hv = hi[v], gv = gg[v];
          float idxf = ((1.0f - ov) * (1.0f - hv) + ov * ((gv > s_g2) ? 1.0f : 0.0f)) * nb;
          if (idxf > 0.0f) { gg[v] = s_g2; op[v] = 1.0f; pp[v] = ind; }
        }
      }
    }
    __syncthreads();
  }

  // outputs: hist copy; path initialized to goal one-hot
  for (int c = 0; c < 16; ++c) {
    int i = (c << 8) + tid;
    out_hist[(size_t)n * HW + i] = hi[i];
    out_path[(size_t)n * HW + i] = goal[(size_t)n * HW + i];
  }
  __syncthreads();
  if (tid == 0) {
    int loc = pp[s_goal];
    for (int t = 0; t < Tmax; ++t) {
      out_path[(size_t)n * HW + loc] = 1.0f;
      loc = pp[loc];
    }
  }
}

// ---------------------------------------------------------------------------
// Host orchestration
// ---------------------------------------------------------------------------
extern "C" void kernel_launch(void* const* d_in, const int* in_sizes, int n_in,
                              void* d_out, int out_size, void* d_ws, size_t ws_size,
                              hipStream_t stream) {
  (void)in_sizes; (void)n_in; (void)out_size; (void)ws_size;

  const float* md = (const float*)d_in[0];
  const float* st = (const float*)d_in[1];
  const float* gl = (const float*)d_in[2];

  // params flattened in sorted-dict-key pytree order: dec, enc, head;
  // each cbr dict -> b, s, w (sorted keys)
  struct Cv { int bi, si, wi, Cin, Cout; };
  Cv dec[8], enc[13];
  const int dCi[8]  = {1024, 256, 512, 128, 256, 64, 64, 32};
  const int dCo[8]  = {256, 256, 128, 128, 64, 64, 32, 32};
  const int eCi[13] = {2, 64, 64, 128, 128, 256, 256, 256, 512, 512, 512, 512, 512};
  const int eCo[13] = {64, 64, 128, 128, 256, 256, 256, 512, 512, 512, 512, 512, 512};
  int p = 3;
  for (int i = 0; i < 8;  ++i) { dec[i] = {p, p + 1, p + 2, dCi[i], dCo[i]}; p += 3; }
  for (int i = 0; i < 13; ++i) { enc[i] = {p, p + 1, p + 2, eCi[i], eCo[i]}; p += 3; }
  int head_b = p, head_w = p + 1;

  // workspace bump allocator (256B aligned)
  char* ws = (char*)d_ws;
  size_t off = 0;
  auto alloc = [&](size_t bytes) -> void* {
    void* q = ws + off;
    off = (off + bytes + 255) & ~(size_t)255;
    return q;
  };

  _Float16* X    = (_Float16*)alloc((size_t)NB * HWFULL * 2 * 2);
  _Float16* tmpA = (_Float16*)alloc((size_t)8388608 * 2);
  _Float16* tmpB = (_Float16*)alloc((size_t)8388608 * 2);
  _Float16* f0   = (_Float16*)alloc((size_t)8388608 * 2);
  _Float16* f1   = (_Float16*)alloc((size_t)4194304 * 2);
  _Float16* f2   = (_Float16*)alloc((size_t)2097152 * 2);
  _Float16* f3   = (_Float16*)alloc((size_t)1048576 * 2);
  _Float16* f4   = (_Float16*)alloc((size_t)262144 * 2);

  _Float16* wtD[8]; _Float16* wtE[13];
  auto kpad_of = [](int Cin) { return ((9 * Cin + 31) / 32) * 32; };
  for (int i = 0; i < 8;  ++i) wtD[i] = (_Float16*)alloc((size_t)dCo[i] * kpad_of(dCi[i]) * 2);
  for (int i = 0; i < 13; ++i) wtE[i] = (_Float16*)alloc((size_t)eCo[i] * kpad_of(eCi[i]) * 2);

  float* costw = (float*)alloc((size_t)NB * HWFULL * 4);
  float* hmap  = (float*)alloc((size_t)NB * HWFULL * 4);
  float* openb = (float*)alloc((size_t)NB * HWFULL * 4);
  float* histb = (float*)alloc((size_t)NB * HWFULL * 4);
  float* gbuf  = (float*)alloc((size_t)NB * HWFULL * 4);
  int*   parb  = (int*)  alloc((size_t)NB * HWFULL * 4);
  int*   gidx  = (int*)  alloc(NB * 4);

  float* out  = (float*)d_out;
  float* out_hist = out;
  float* out_path = out + NB * HWFULL;
  float* out_cost = out + 2 * NB * HWFULL;

  // weight transforms
  auto xform = [&](const Cv& c, _Float16* wt) {
    int K = 9 * c.Cin, Kp = kpad_of(c.Cin);
    int tot = c.Cout * Kp;
    xform_w_k<<<ceil_div(tot, 256), 256, 0, stream>>>(
        (const float*)d_in[c.wi], wt, c.Cin, c.Cout, K, Kp);
  };
  for (int i = 0; i < 8;  ++i) xform(dec[i], wtD[i]);
  for (int i = 0; i < 13; ++i) xform(enc[i], wtE[i]);

  // conv launcher: 2D grid (m-groups of 8 mtiles, n-groups of NT ntiles)
  auto conv = [&](const _Float16* in, _Float16* o, const Cv& c, _Float16* wt, int H) {
    int W = H, HWl = H * W;
    int logW = __builtin_ctz(W), logHW = __builtin_ctz(HWl);
    int K = 9 * c.Cin, Kp = kpad_of(c.Cin);
    int logCin = __builtin_ctz(c.Cin);
    int M = NB * HWl;
    if (c.Cout % 64 == 0) {
      dim3 grid(M / 128, c.Cout / 64);
      conv_wmma_k<4><<<grid, 256, 0, stream>>>(
          in, wt, (const float*)d_in[c.si], (const float*)d_in[c.bi], o,
          H, W, logW, logHW, c.Cin, logCin, c.Cout, K, Kp, 1);
    } else {  // Cout == 32
      dim3 grid(M / 128, c.Cout / 32);
      conv_wmma_k<2><<<grid, 256, 0, stream>>>(
          in, wt, (const float*)d_in[c.si], (const float*)d_in[c.bi], o,
          H, W, logW, logHW, c.Cin, logCin, c.Cout, K, Kp, 1);
    }
  };
  auto pool = [&](const _Float16* in, _Float16* o, int H, int C) {
    int tot = NB * (H >> 1) * (H >> 1) * C;
    maxpool2_k<<<ceil_div(tot, 256), 256, 0, stream>>>(in, o, H, C);
  };
  auto upcat = [&](const _Float16* a, const _Float16* sk, _Float16* o, int Ho, int C1, int C2) {
    int tot = NB * Ho * Ho * (C1 + C2);
    up2cat_k<<<ceil_div(tot, 256), 256, 0, stream>>>(a, sk, o, Ho, C1, C2);
  };

  // ---- UNet forward ----
  prep_input_k<<<ceil_div(NB * HWFULL, 256), 256, 0, stream>>>(md, st, gl, X);

  conv(X,    tmpA, enc[0], wtE[0], 64);                 // 2->64
  conv(tmpA, f0,   enc[1], wtE[1], 64);                 // 64->64
  pool(f0, tmpA, 64, 64);
  conv(tmpA, tmpB, enc[2], wtE[2], 32);                 // 64->128
  conv(tmpB, f1,   enc[3], wtE[3], 32);
  pool(f1, tmpA, 32, 128);
  conv(tmpA, tmpB, enc[4], wtE[4], 16);                 // 128->256
  conv(tmpB, tmpA, enc[5], wtE[5], 16);
  conv(tmpA, f2,   enc[6], wtE[6], 16);
  pool(f2, tmpA, 16, 256);
  conv(tmpA, tmpB, enc[7], wtE[7], 8);                  // 256->512
  conv(tmpB, tmpA, enc[8], wtE[8], 8);
  conv(tmpA, f3,   enc[9], wtE[9], 8);
  pool(f3, tmpA, 8, 512);
  conv(tmpA, tmpB, enc[10], wtE[10], 4);
  conv(tmpB, tmpA, enc[11], wtE[11], 4);
  conv(tmpA, f4,   enc[12], wtE[12], 4);

  upcat(f4, f3, tmpA, 8, 512, 512);                     // -> 1024@8
  conv(tmpA, tmpB, dec[0], wtD[0], 8);                  // 1024->256
  conv(tmpB, tmpA, dec[1], wtD[1], 8);
  upcat(tmpA, f2, tmpB, 16, 256, 256);                  // -> 512@16
  conv(tmpB, tmpA, dec[2], wtD[2], 16);                 // 512->128
  conv(tmpA, tmpB, dec[3], wtD[3], 16);
  upcat(tmpB, f1, tmpA, 32, 128, 128);                  // -> 256@32
  conv(tmpA, tmpB, dec[4], wtD[4], 32);                 // 256->64
  conv(tmpB, tmpA, dec[5], wtD[5], 32);
  upcat(tmpA, (const _Float16*)nullptr, tmpB, 64, 64, 0); // -> 64@64
  conv(tmpB, tmpA, dec[6], wtD[6], 64);                 // 64->32
  conv(tmpA, tmpB, dec[7], wtD[7], 64);                 // 32->32

  head_k<<<ceil_div(NB * HWFULL, 256), 256, 0, stream>>>(
      tmpB, (const float*)d_in[head_w], (const float*)d_in[head_b], costw, out_cost);

  // ---- A* ----
  find_goal_k<<<ceil_div(NB * HWFULL, 256), 256, 0, stream>>>(gl, gidx);
  compute_h_k<<<ceil_div(NB * HWFULL, 256), 256, 0, stream>>>(costw, gidx, hmap);
  init_astar_k<<<ceil_div(NB * HWFULL, 256), 256, 0, stream>>>(st, gidx, openb, histb, gbuf, parb);
  astar_k<<<NB, 256, 0, stream>>>(costw, hmap, md, gl, gidx,
                                  openb, histb, gbuf, parb,
                                  out_hist, out_path, 1024);
}
